// Moe_13305808683088
// MI455X (gfx1250) — matrix operand order repaired and checked by reading, not compiled, not run on hardware
//
#include <hip/hip_runtime.h>
#include <hip/hip_bf16.h>
#include <math.h>

// ---------------------------------------------------------------------------
// MoE pipeline for MI455X (gfx1250, wave32).
// Heavy math: v_wmma_f32_16x16x32_bf16 (f32 accumulate).
// A-tile staging: Tensor Data Mover (tensor_load_to_lds, TENSORcnt).
// B-tile staging: global_load_async_to_lds_b128 (ASYNCcnt).
// Double-buffered LDS; all B operands pre-transposed to [N][K].
// ---------------------------------------------------------------------------

typedef __attribute__((ext_vector_type(16))) __bf16 v16bf;
typedef __attribute__((ext_vector_type(8)))  __bf16 v8bf;
typedef __attribute__((ext_vector_type(8)))  float  v8f;
typedef __attribute__((ext_vector_type(4)))  unsigned int v4u;
typedef __attribute__((ext_vector_type(4)))  int v4i;
typedef __attribute__((ext_vector_type(8)))  int v8i;

#if __has_builtin(__builtin_amdgcn_tensor_load_to_lds)
#define HAVE_TDM 1
#else
#define HAVE_TDM 0
#endif

__device__ __forceinline__ __bf16 f2bf(float f) {
  unsigned int u = __builtin_bit_cast(unsigned int, f);
  unsigned int r = (u + 0x7FFFu + ((u >> 16) & 1u)) >> 16;
  unsigned short s = (unsigned short)r;
  return __builtin_bit_cast(__bf16, s);
}
__device__ __forceinline__ float bf2f(__bf16 b) {
  unsigned short s = __builtin_bit_cast(unsigned short, b);
  unsigned int u = ((unsigned int)s) << 16;
  return __builtin_bit_cast(float, u);
}

// Async DMA: 16 bytes per lane, global -> LDS, tracked by ASYNCcnt.
// LDS address = low 32 bits of the generic shared pointer (aperture rule).
__device__ __forceinline__ void async_copy_b128(unsigned lds_off, const void* g) {
  asm volatile("global_load_async_to_lds_b128 %0, %1, off"
               :: "v"(lds_off), "v"((unsigned long long)(uintptr_t)g)
               : "memory");
}

template <int N>
__device__ __forceinline__ void wait_async() {
#if __has_builtin(__builtin_amdgcn_s_wait_asynccnt)
  __builtin_amdgcn_s_wait_asynccnt((short)N);
#else
  if constexpr (N == 0)      asm volatile("s_wait_asynccnt 0" ::: "memory");
  else if constexpr (N == 2) asm volatile("s_wait_asynccnt 2" ::: "memory");
  else                       asm volatile("s_wait_asynccnt 6" ::: "memory");
#endif
}

template <int N>
__device__ __forceinline__ void wait_tensor() {
#if __has_builtin(__builtin_amdgcn_s_wait_tensorcnt)
  __builtin_amdgcn_s_wait_tensorcnt((short)N);
#else
  if constexpr (N == 0) asm volatile("s_wait_tensorcnt 0" ::: "memory");
  else                  asm volatile("s_wait_tensorcnt 1" ::: "memory");
#endif
}

#if HAVE_TDM
// TDM 2-D tile load: tileRows x tileK bf16 elements, row stride rowLen
// (elements), global tile start gaddr -> compact row-major tile at lds_off.
// D# fields per CDNA5 ISA 8.3/8.4 (group0: count/lds/global/type,
// group1: data_size=2B, tensor dims, tile dims, dim0 stride).
__device__ __forceinline__ void tdm_load_2d(unsigned lds_off, const void* gaddr,
                                            int rowLen, int numRows,
                                            int tileK, int tileRows) {
  unsigned long long ga = (unsigned long long)(uintptr_t)gaddr;
  v4u g0;
  g0[0] = 0x1u;                                  // count=1, user descriptor
  g0[1] = lds_off;                               // lds_addr (bytes)
  g0[2] = (unsigned)(ga & 0xFFFFFFFFu);          // global_addr[31:0]
  g0[3] = (unsigned)((ga >> 32) & 0x01FFFFFFu)   // global_addr[56:32]
        | (2u << 30);                            // type = 2 ("image")
  v8i g1;
  g1[0] = (1 << 16);                             // wg_mask=0, data_size=2B
  g1[1] = (int)(((unsigned)rowLen & 0xFFFFu) << 16);            // dim0 lo16
  g1[2] = (int)((((unsigned)rowLen >> 16) & 0xFFFFu)            // dim0 hi16
        | (((unsigned)numRows & 0xFFFFu) << 16));               // dim1 lo16
  g1[3] = (int)((((unsigned)numRows >> 16) & 0xFFFFu)           // dim1 hi16
        | ((unsigned)tileK << 16));                             // tile_dim0
  g1[4] = tileRows;                              // tile_dim1 (tile_dim2 = 0)
  g1[5] = rowLen;                                // tensor_dim0_stride lo32
  g1[6] = 0;
  g1[7] = 0;
  v4i z4 = {0, 0, 0, 0};
#if defined(__clang_major__) && (__clang_major__ >= 23)
  v8i z8 = {0, 0, 0, 0, 0, 0, 0, 0};
  __builtin_amdgcn_tensor_load_to_lds(g0, g1, z4, z4, z8, 0);
#else
  __builtin_amdgcn_tensor_load_to_lds(g0, g1, z4, z4, 0);
#endif
}
#endif

// ---- A fragment: 16x32 bf16, row M = lane%16.
// lanes 0-15 hold K = {0..7, 16..23}; lanes 16-31 hold K = {8..15, 24..31}.
__device__ __forceinline__ v16bf load_frag_a(const __bf16* lds, int row, int kb) {
  const v8bf lo = *(const v8bf*)(lds + row * 32 + kb);
  const v8bf hi = *(const v8bf*)(lds + row * 32 + kb + 16);
  v16bf r;
#pragma unroll
  for (int i = 0; i < 8; ++i) { r[i] = lo[i]; r[i + 8] = hi[i]; }
  return r;
}

// ---- B fragment: 32x16 bf16 (KxN), col N = lane%16; LDS holds B^T [n][k].
__device__ __forceinline__ v16bf load_frag_b(const __bf16* lds, int col, int kb) {
  const v8bf lo = *(const v8bf*)(lds + col * 32 + kb);
  const v8bf hi = *(const v8bf*)(lds + col * 32 + kb + 16);
  v16bf r;
#pragma unroll
  for (int i = 0; i < 8; ++i) { r[i] = lo[i]; r[i + 8] = hi[i]; }
  return r;
}

// ---------------------------------------------------------------------------
// Batched bf16 GEMM: C[M,N] = A[M,K] * B[K,N] (+bias[N]) (+relu)
// A row-major [M][K]; BT row-major [N][K]. Out f32/bf16, opt. transposed.
// M % 256 == 0, N % 128 == 0, K % 32 == 0.
// 256 threads = 8 waves (4x2); WG tile 256x128; wave tile 64x64 (4x4 WMMA).
// ---------------------------------------------------------------------------
template <bool OUT_BF16, bool OUT_T, bool RELU, bool HAS_BIAS>
__global__ __launch_bounds__(256) void gemm_bf16_wmma(
    const __bf16* __restrict__ A, const __bf16* __restrict__ BT,
    const float* __restrict__ bias, float* __restrict__ Cf,
    __bf16* __restrict__ Cb, int M, int N, int K, int ldOut,
    long strideA, long strideB, long strideBias, long strideC) {
  __shared__ __bf16 sA[2][256 * 32];
  __shared__ __bf16 sB[2][128 * 32];

  const int batch = blockIdx.z;
  A  += (long)batch * strideA;
  BT += (long)batch * strideB;
  if (HAS_BIAS) bias += (long)batch * strideBias;
  if (OUT_BF16) Cb += (long)batch * strideC; else Cf += (long)batch * strideC;

  const int tid  = threadIdx.x;
  const int wave = tid >> 5;
  const int lane = tid & 31;
  const int l16  = lane & 15;
  const int m0 = blockIdx.y * 256;
  const int n0 = blockIdx.x * 128;
  const int wr = wave >> 1;  // 0..3 : 64-row slab
  const int wc = wave & 1;   // 0..1 : 64-col slab

  const int kbA = (lane < 16) ? 0 : 8;
  const int kbB = (lane < 16) ? 0 : 8;

  const int arow = tid;             // fallback A staging: one 64B row/thread
  const int brow = tid >> 1;        // B staging: half row (32B) per thread
  const int bseg = (tid & 1) * 16;

  v8f acc[4][4] = {};

  // Stage one 256x32 A tile + 128x32 B^T tile into LDS buffer `buf`.
  auto stage = [&](int k0, int buf) {
#if HAVE_TDM
    if (wave == 0)  // one TDM descriptor covers the whole A tile
      tdm_load_2d((unsigned)(uintptr_t)&sA[buf][0],
                  A + (long)m0 * K + k0, K, M, 32, 256);
#else
    {
      const __bf16* ga = A + (long)(m0 + arow) * K + k0;
      unsigned la = (unsigned)(uintptr_t)&sA[buf][arow * 32];
      async_copy_b128(la,      ga);
      async_copy_b128(la + 16, ga + 8);
      async_copy_b128(la + 32, ga + 16);
      async_copy_b128(la + 48, ga + 24);
    }
#endif
    const __bf16* gb = BT + (long)(n0 + brow) * K + k0 + bseg;
    unsigned lb = (unsigned)(uintptr_t)&sB[buf][brow * 32 + bseg];
    async_copy_b128(lb,      gb);
    async_copy_b128(lb + 16, gb + 8);
  };

  auto compute = [&](int buf) {
    const __bf16* sAc = sA[buf];
    const __bf16* sBc = sB[buf];
    v16bf bfrag[4];
#pragma unroll
    for (int ct = 0; ct < 4; ++ct)
      bfrag[ct] = load_frag_b(sBc, wc * 64 + ct * 16 + l16, kbB);
#pragma unroll
    for (int rt = 0; rt < 4; ++rt) {
      v16bf afrag = load_frag_a(sAc, wr * 64 + rt * 16 + l16, kbA);
#pragma unroll
      for (int ct = 0; ct < 4; ++ct)
        acc[rt][ct] = __builtin_amdgcn_wmma_f32_16x16x32_bf16(
            false, afrag, false, bfrag[ct], (short)0, acc[rt][ct], false, false);
    }
  };

  constexpr int ASYNC_STEADY = HAVE_TDM ? 2 : 6;

  stage(0, 0);
  int cur = 0;
  int k0 = 0;
  for (; k0 + 32 < K; k0 += 32) {      // steady state: next tile in flight
    stage(k0 + 32, cur ^ 1);
#if HAVE_TDM
    if (wave == 0) wait_tensor<1>();   // current A tile resident (in-order)
#endif
    wait_async<ASYNC_STEADY>();        // current B tile resident
    __syncthreads();
    compute(cur);
    __syncthreads();                   // all waves done reading buffer `cur`
    cur ^= 1;
  }
#if HAVE_TDM
  if (wave == 0) wait_tensor<0>();
#endif
  wait_async<0>();
  __syncthreads();
  compute(cur);                        // last tile, no further staging

  // ---- writeback. C/D layout: col = lane%16, rows (lane<16 ? 0..7 : 8..15)
  const int rbase = (lane < 16) ? 0 : 8;
#pragma unroll
  for (int rt = 0; rt < 4; ++rt) {
#pragma unroll
    for (int ct = 0; ct < 4; ++ct) {
      const int col = n0 + wc * 64 + ct * 16 + l16;
      const float bv = HAS_BIAS ? bias[col] : 0.f;
#pragma unroll
      for (int r = 0; r < 8; ++r) {
        const int rowg = m0 + wr * 64 + rt * 16 + rbase + r;
        float v = acc[rt][ct][r] + bv;
        if (RELU) v = v > 0.f ? v : 0.f;
        const long idx = OUT_T ? ((long)col * ldOut + rowg)
                               : ((long)rowg * N + col);
        if constexpr (OUT_BF16) Cb[idx] = f2bf(v);
        else                    Cf[idx] = v;
      }
    }
  }
}

// ---------------------------------------------------------------------------
// Conversion / padding / transpose kernels (single bandwidth pass each)
// ---------------------------------------------------------------------------
__global__ void cvt_f32_bf16(const float* __restrict__ in,
                             __bf16* __restrict__ out, long n) {
  long i = (long)blockIdx.x * blockDim.x + threadIdx.x;
  if (i < n) out[i] = f2bf(in[i]);
}

// x [M][D] -> xT bf16 [D][M]
__global__ void cvt_xT(const float* __restrict__ in, __bf16* __restrict__ out,
                       int M, int D) {
  long i = (long)blockIdx.x * blockDim.x + threadIdx.x;
  if (i >= (long)M * D) return;
  int m = (int)(i % M);
  int d = (int)(i / M);
  out[i] = f2bf(in[(long)m * D + d]);
}

// phi [D][NP] -> phiT bf16 [NP][D]
__global__ void cvt_phiT(const float* __restrict__ in, __bf16* __restrict__ out,
                         int D, int NP) {
  long i = (long)blockIdx.x * blockDim.x + threadIdx.x;
  if (i >= (long)D * NP) return;
  int d = (int)(i % D);
  int np = (int)(i / D);
  out[i] = f2bf(in[(long)d * NP + np]);
}

// W1 [16][1024][1365] -> W1T bf16 [16][1408][1024] (pad h with zeros)
__global__ void cvt_w1T(const float* __restrict__ in, __bf16* __restrict__ out) {
  long i = (long)blockIdx.x * blockDim.x + threadIdx.x;
  const long total = 16L * 1408 * 1024;
  if (i >= total) return;
  int d = (int)(i % 1024);
  long rem = i / 1024;
  int hp = (int)(rem % 1408);
  int n = (int)(rem / 1408);
  out[i] = (hp < 1365) ? f2bf(in[((long)n * 1024 + d) * 1365 + hp])
                       : __builtin_bit_cast(__bf16, (unsigned short)0);
}

// W2 [16][1365][1024] -> W2T bf16 [16][1024][1408] (pad h with zeros)
__global__ void cvt_w2T(const float* __restrict__ in, __bf16* __restrict__ out) {
  long i = (long)blockIdx.x * blockDim.x + threadIdx.x;
  const long total = 16L * 1024 * 1408;
  if (i >= total) return;
  int hp = (int)(i % 1408);
  long rem = i / 1408;
  int o = (int)(rem % 1024);
  int n = (int)(rem / 1024);
  out[i] = (hp < 1365) ? f2bf(in[((long)n * 1365 + hp) * 1024 + o])
                       : __builtin_bit_cast(__bf16, (unsigned short)0);
}

// b1 [16][1365] -> f32 [16][1408] zero-padded
__global__ void cvt_pad_b1(const float* __restrict__ in, float* __restrict__ out) {
  int i = blockIdx.x * blockDim.x + threadIdx.x;
  if (i >= 16 * 1408) return;
  int hp = i % 1408;
  int n = i / 1408;
  out[i] = (hp < 1365) ? in[n * 1365 + hp] : 0.f;
}

// ---------------------------------------------------------------------------
// Softmax over experts (axis n) for each (m, p); writes D^T [np][m] bf16.
// ---------------------------------------------------------------------------
__global__ void softmax_experts(const __bf16* __restrict__ lg,
                                __bf16* __restrict__ DT, int M) {
  int idx = blockIdx.x * blockDim.x + threadIdx.x;  // m*1024 + p
  int m = idx >> 10;
  int p = idx & 1023;
  if (m >= M) return;
  float v[16];
  float mx = -3.0e38f;
#pragma unroll
  for (int n = 0; n < 16; ++n) {
    v[n] = bf2f(lg[(long)m * 16384 + n * 1024 + p]);
    mx = fmaxf(mx, v[n]);
  }
  float sum = 0.f;
#pragma unroll
  for (int n = 0; n < 16; ++n) { v[n] = __expf(v[n] - mx); sum += v[n]; }
  float inv = 1.f / sum;
#pragma unroll
  for (int n = 0; n < 16; ++n)
    DT[(long)(n * 1024 + p) * M + m] = f2bf(v[n] * inv);
}

// ---------------------------------------------------------------------------
// Softmax over all 16384 (expert,slot) pairs per token; 1 block per row.
// ---------------------------------------------------------------------------
__global__ __launch_bounds__(256) void softmax_token(
    const __bf16* __restrict__ lg, __bf16* __restrict__ Cm) {
  const int m = blockIdx.x;
  const int tid = threadIdx.x;
  const __bf16* row = lg + (long)m * 16384;
  __shared__ float red[256];

  float mx = -3.0e38f;
  for (int i = tid; i < 16384; i += 256) mx = fmaxf(mx, bf2f(row[i]));
  red[tid] = mx;
  __syncthreads();
  for (int s = 128; s > 0; s >>= 1) {
    if (tid < s) red[tid] = fmaxf(red[tid], red[tid + s]);
    __syncthreads();
  }
  mx = red[0];
  __syncthreads();

  float sum = 0.f;
  for (int i = tid; i < 16384; i += 256) sum += __expf(bf2f(row[i]) - mx);
  red[tid] = sum;
  __syncthreads();
  for (int s = 128; s > 0; s >>= 1) {
    if (tid < s) red[tid] += red[tid + s];
    __syncthreads();
  }
  const float inv = 1.f / red[0];

  for (int i = tid; i < 16384; i += 256)
    Cm[(long)m * 16384 + i] = f2bf(__expf(bf2f(row[i]) - mx) * inv);
}

// ---------------------------------------------------------------------------
extern "C" void kernel_launch(void* const* d_in, const int* in_sizes, int n_in,
                              void* d_out, int out_size, void* d_ws,
                              size_t ws_size, hipStream_t stream) {
  const float* x   = (const float*)d_in[0];  // [4096,1024]
  const float* phi = (const float*)d_in[1];  // [1024,16,1024] == [1024,16384]
  const float* W1  = (const float*)d_in[2];  // [16,1024,1365]
  const float* b1  = (const float*)d_in[3];  // [16,1365]
  const float* W2  = (const float*)d_in[4];  // [16,1365,1024]
  const float* b2  = (const float*)d_in[5];  // [16,1024]

  constexpr int M = 4096, D = 1024, NE = 16, P = 1024, HP = 1408;
  constexpr int NP = NE * P;  // 16384

  char* ws = (char*)d_ws;
  size_t off = 0;
  auto take = [&](size_t bytes) -> char* {
    char* p = ws + off;
    off += (bytes + 255) & ~(size_t)255;
    return p;
  };

  __bf16* xb   = (__bf16*)take((size_t)M * D * 2);        // x, row-major
  __bf16* xbT  = (__bf16*)take((size_t)D * M * 2);        // x^T
  __bf16* phiT = (__bf16*)take((size_t)NP * D * 2);       // phi^T
  __bf16* W1T  = (__bf16*)take((size_t)NE * HP * D * 2);  // W1^T (padded)
  __bf16* W2T  = (__bf16*)take((size_t)NE * P * HP * 2);  // W2^T (padded)
  float*  b1p  = (float*) take((size_t)NE * HP * 4);
  __bf16* lg   = (__bf16*)take((size_t)M * NP * 2);       // logits
  __bf16* DTm  = (__bf16*)take((size_t)NP * M * 2);       // dispatch^T
  __bf16* Cm   = (__bf16*)take((size_t)M * NP * 2);       // combine
  __bf16* Xs   = (__bf16*)take((size_t)NP * D * 2);       // slot inputs
  __bf16* Hs   = (__bf16*)take((size_t)NE * P * HP * 2);  // hidden
  __bf16* YsT  = (__bf16*)take((size_t)P * NP * 2);       // Ys^T [o][np]

  // --- precision conversion / padding / transposes
  cvt_f32_bf16<<<(M * D) / 256, 256, 0, stream>>>(x, xb, (long)M * D);
  cvt_xT      <<<(M * D) / 256, 256, 0, stream>>>(x, xbT, M, D);
  cvt_phiT    <<<(D * NP) / 256, 256, 0, stream>>>(phi, phiT, D, NP);
  cvt_w1T     <<<(NE * HP * D) / 256, 256, 0, stream>>>(W1, W1T);
  cvt_w2T     <<<(NE * P * HP) / 256, 256, 0, stream>>>(W2, W2T);
  cvt_pad_b1  <<<(NE * HP + 255) / 256, 256, 0, stream>>>(b1, b1p);

  // --- 1) logits = x * phi  [4096 x 16384], K=1024
  gemm_bf16_wmma<true, false, false, false>
      <<<dim3(NP / 128, M / 256, 1), 256, 0, stream>>>(
      xb, phiT, nullptr, nullptr, lg, M, NP, D, 0, 0, 0, 0, 0);

  // --- 2) dispatch softmax -> D^T ; combine softmax -> C
  softmax_experts<<<(M * P) / 256, 256, 0, stream>>>(lg, DTm, M);
  softmax_token<<<M, 256, 0, stream>>>(lg, Cm);

  // --- 3) Xs = D^T * x  [16384 x 1024], K=4096  (B^T = x^T)
  gemm_bf16_wmma<true, false, false, false>
      <<<dim3(D / 128, NP / 256, 1), 256, 0, stream>>>(
      DTm, xbT, nullptr, nullptr, Xs, NP, D, M, 0, 0, 0, 0, 0);

  // --- 4) H = relu(Xs*W1 + b1), 16 experts, [1024 x 1408], K=1024
  gemm_bf16_wmma<true, false, true, true>
      <<<dim3(HP / 128, P / 256, NE), 256, 0, stream>>>(
      Xs, W1T, b1p, nullptr, Hs, P, HP, D, 0,
      (long)P * D, (long)HP * D, (long)HP, (long)P * HP);

  // --- 5) Ys = H*W2 + b2, 16 experts, [1024 x 1024], K=1408;
  //         written transposed -> YsT[o][n*1024+p]
  gemm_bf16_wmma<true, true, false, true>
      <<<dim3(P / 128, P / 256, NE), 256, 0, stream>>>(
      Hs, W2T, b2, nullptr, YsT, P, P, HP, NP,
      (long)P * HP, (long)P * HP, (long)P, (long)P);

  // --- 6) Y = C * Ys  [4096 x 1024], K=16384 -> f32 output (B^T = YsT)
  gemm_bf16_wmma<false, false, false, false>
      <<<dim3(P / 128, M / 256, 1), 256, 0, stream>>>(
      Cm, YsT, nullptr, (float*)d_out, nullptr, M, P, NP, 0, 0, 0, 0, 0);
}
